// MSA_BLOCK_15281493639230
// MI455X (gfx1250) — matrix operand order repaired and checked by reading, not compiled, not run on hardware
//
#include <hip/hip_runtime.h>

// MSA block: q,k,v = x@W^T ; scores = q k^T * DH^-0.5 + bias ; softmax ; @v ; out-proj.
// B=8, N=1024, D=512, H=8, DH=64.
// - All matmuls: v_wmma_f32_16x16x32_bf16 (wave32, fp32 accumulate).
// - Flash online softmax: the 256MB bias is the only O(N^2) HBM stream, read once.
// - All 8 heads of a (b, 16-row) tile share one 8-wave workgroup; the bias tile is
//   staged into LDS (async global->LDS, double buffered) so each 128B line is fully
//   consumed inside one WGP instead of 1/8-used by 8 different WGPs.

#define BB   8
#define NN   1024
#define DDIM 512
#define HH   8
#define DH   64
#define PPITCH 40   // P-tile LDS pitch (ushorts), multiple of 8 for 16B-aligned reads
#define BPITCH 264  // bias tile LDS pitch (floats): 256 + 8 pad

#define USE_ASYNC_BIAS 1  // global_load_async_to_lds_b128 path (fallback: ld+ds_store)

typedef __attribute__((ext_vector_type(16))) __bf16 v16bf;
typedef __attribute__((ext_vector_type(8)))  float  v8f;

union Frag {
  uint4 q[2];
  v16bf v;
  unsigned short u[16];
};

__device__ __forceinline__ unsigned short f2bf(float f) {
  unsigned u = __float_as_uint(f);
  unsigned r = u + 0x7FFFu + ((u >> 16) & 1u);  // round-to-nearest-even
  return (unsigned short)(r >> 16);
}

// ---------------------------------------------------------------- convert
__global__ void f32_to_bf16_kernel(const float* __restrict__ src,
                                   unsigned short* __restrict__ dst, int n) {
  int i = blockIdx.x * blockDim.x + threadIdx.x;
  int stride = gridDim.x * blockDim.x;
  for (; i < n; i += stride) dst[i] = f2bf(src[i]);
}

// ---------------------------------------------------------------- QKV GEMM
// One wave computes a 16x64 tile of (x @ W^T), k-dim double-buffered.
__global__ __launch_bounds__(32) void qkv_gemm_kernel(
    const unsigned short* __restrict__ xb,
    const unsigned short* __restrict__ wqb,
    const unsigned short* __restrict__ wkb,
    const unsigned short* __restrict__ wvb,
    unsigned short* __restrict__ qb,   // [B,H,N,DH]
    unsigned short* __restrict__ kb,   // [B,H,N,DH]
    unsigned short* __restrict__ vb) { // [B,H,DH,N] (transposed for PV B-frags)
  const int m0    = blockIdx.x * 16;
  const int n0    = blockIdx.y * 64;
  const int which = blockIdx.z;  // 0=q 1=k 2=v
  const unsigned short* W = (which == 0) ? wqb : (which == 1) ? wkb : wvb;
  const int lane = threadIdx.x;
  const int col  = lane & 15;
  const int hlf  = lane >> 4;

  const unsigned short* xrow = xb + (size_t)(m0 + col) * DDIM + hlf * 8;
  const unsigned short* wrow0 = W + (size_t)(n0 + col) * DDIM + hlf * 8;

  v8f acc[4];
#pragma unroll
  for (int t = 0; t < 4; ++t)
#pragma unroll
    for (int r = 0; r < 8; ++r) acc[t][r] = 0.0f;

  Frag a[2], b[2][4];
  auto loadA = [&](Frag& f, int k0) {
    f.q[0] = *(const uint4*)(xrow + k0);
    f.q[1] = *(const uint4*)(xrow + k0 + 16);
  };
  auto loadB = [&](Frag* f, int k0) {
#pragma unroll
    for (int t = 0; t < 4; ++t) {
      const unsigned short* wr = wrow0 + (size_t)t * 16 * DDIM + k0;
      f[t].q[0] = *(const uint4*)(wr);
      f[t].q[1] = *(const uint4*)(wr + 16);
    }
  };

  loadA(a[0], 0);
  loadB(b[0], 0);
  for (int k0 = 0; k0 < DDIM; k0 += 64) {
    loadA(a[1], k0 + 32);
    loadB(b[1], k0 + 32);
#pragma unroll
    for (int t = 0; t < 4; ++t)
      acc[t] = __builtin_amdgcn_wmma_f32_16x16x32_bf16(
          false, a[0].v, false, b[0][t].v, (short)0, acc[t], false, false);
    if (k0 + 64 < DDIM) {
      loadA(a[0], k0 + 64);
      loadB(b[0], k0 + 64);
    }
#pragma unroll
    for (int t = 0; t < 4; ++t)
      acc[t] = __builtin_amdgcn_wmma_f32_16x16x32_bf16(
          false, a[1].v, false, b[1][t].v, (short)0, acc[t], false, false);
  }

  // C/D layout: element r, lane -> row m0 + r + hlf*8, col n0 + t*16 + lane%16
#pragma unroll
  for (int t = 0; t < 4; ++t) {
#pragma unroll
    for (int r = 0; r < 8; ++r) {
      const int i  = m0 + r + hlf * 8;
      const int j  = n0 + t * 16 + col;
      const int bi = i >> 10, n = i & 1023;
      const int hh = j >> 6,  dh = j & 63;
      const float val = acc[t][r];
      if (which == 0) {
        qb[(((size_t)bi * HH + hh) * NN + n) * DH + dh] = f2bf(val * 0.125f);
      } else if (which == 1) {
        kb[(((size_t)bi * HH + hh) * NN + n) * DH + dh] = f2bf(val);
      } else {
        vb[(((size_t)bi * HH + hh) * DH + dh) * NN + n] = f2bf(val);
      }
    }
  }
}

// ---------------------------------------------------------------- attention
// Workgroup = 8 waves = 8 heads for one (b, 16-query-row) tile. Bias tile
// [16 x 32 x 8h] staged to LDS (double buffered, async) with coalesced B128s.
__global__ __launch_bounds__(256) void attn_kernel(
    const unsigned short* __restrict__ qg,   // [B,H,N,DH]
    const unsigned short* __restrict__ kg,   // [B,H,N,DH]
    const unsigned short* __restrict__ vg,   // [B,H,DH,N]
    const float* __restrict__ bias,          // [B,N,N,H]
    unsigned short* __restrict__ aob) {      // [B,N,D] bf16
  const int i0 = blockIdx.x * 16;
  const int b  = blockIdx.y;
  const int h    = threadIdx.y;  // wave == head
  const int lane = threadIdx.x;
  const int col  = lane & 15;
  const int hlf  = lane >> 4;
  const int tid  = h * 32 + lane;  // 0..255

  const unsigned short* qh = qg + ((size_t)(b * HH + h) * NN) * DH;
  const unsigned short* kh = kg + ((size_t)(b * HH + h) * NN) * DH;
  const unsigned short* vh = vg + ((size_t)(b * HH + h) * DH) * NN;
  const float* biasb = bias + ((size_t)(b * NN + i0) * NN) * HH;

  __shared__ float bias_tile[2][16 * BPITCH];          // 2 x 16.5 KB
  __shared__ unsigned short pshare[HH][16 * PPITCH];   // per-wave P transpose

  // Cooperative bias staging: 16KB tile = 1024 float4; 4 per thread.
  auto stage_bias = [&](int j0, int buf) {
    float* dst = &bias_tile[buf][0];
#pragma unroll
    for (int q2 = 0; q2 < 4; ++q2) {
      const int flat = tid + 256 * q2;
      const int i = flat >> 6, c = flat & 63;  // row, float4-chunk
      const float* g = biasb + ((size_t)i * NN + j0) * HH + c * 4;
      float* l = dst + i * BPITCH + c * 4;
#if USE_ASYNC_BIAS
      const unsigned lds_off = (unsigned)(size_t)l;            // LDS byte offset
      const unsigned long long ga = (unsigned long long)(size_t)g;
      asm volatile("global_load_async_to_lds_b128 %0, %1, off"
                   :: "v"(lds_off), "v"(ga) : "memory");
#else
      *(float4*)l = *(const float4*)g;
#endif
    }
  };
  auto stage_wait = [&]() {
#if USE_ASYNC_BIAS
    asm volatile("s_wait_asynccnt 0x0" ::: "memory");
#endif
    __syncthreads();
  };

  // Q A-fragments for K = 0..31 and 32..63
  Frag aq[2];
  {
    const unsigned short* qrow = qh + (size_t)(i0 + col) * DH + hlf * 8;
    aq[0].q[0] = *(const uint4*)(qrow);
    aq[0].q[1] = *(const uint4*)(qrow + 16);
    aq[1].q[0] = *(const uint4*)(qrow + 32);
    aq[1].q[1] = *(const uint4*)(qrow + 48);
  }

  float mrun[8], lrun[8];
  v8f oacc[4];
#pragma unroll
  for (int r = 0; r < 8; ++r) { mrun[r] = -1e30f; lrun[r] = 0.0f; }
#pragma unroll
  for (int t = 0; t < 4; ++t)
#pragma unroll
    for (int r = 0; r < 8; ++r) oacc[t][r] = 0.0f;

  int cur = 0;
  stage_bias(0, 0);
  stage_wait();

  for (int j0 = 0; j0 < NN; j0 += 32) {
    if (j0 + 32 < NN) stage_bias(j0 + 32, cur ^ 1);  // overlap next-tile stage
    if (j0 + 64 < NN)                                // prefetch tile after that
      __builtin_prefetch(biasb + ((size_t)(tid >> 6) * NN + j0 + 64) * HH +
                             (tid & 63) * 4, 0, 0);

    // ---- scores: two 16x16 tiles (keys j0..+15, j0+16..+31)
    v8f s0, s1;
#pragma unroll
    for (int r = 0; r < 8; ++r) { s0[r] = 0.0f; s1[r] = 0.0f; }
#pragma unroll
    for (int kk = 0; kk < 2; ++kk) {
      Frag bk;
      const unsigned short* krow = kh + (size_t)(j0 + col) * DH + kk * 32 + hlf * 8;
      bk.q[0] = *(const uint4*)(krow);
      bk.q[1] = *(const uint4*)(krow + 16);
      s0 = __builtin_amdgcn_wmma_f32_16x16x32_bf16(
          false, aq[kk].v, false, bk.v, (short)0, s0, false, false);
    }
#pragma unroll
    for (int kk = 0; kk < 2; ++kk) {
      Frag bk;
      const unsigned short* krow =
          kh + (size_t)(j0 + 16 + col) * DH + kk * 32 + hlf * 8;
      bk.q[0] = *(const uint4*)(krow);
      bk.q[1] = *(const uint4*)(krow + 16);
      s1 = __builtin_amdgcn_wmma_f32_16x16x32_bf16(
          false, aq[kk].v, false, bk.v, (short)0, s1, false, false);
    }

    // ---- add bias from the staged LDS tile (conflict-free: bank = j*8+h)
    {
      const float* bt = &bias_tile[cur][0];
#pragma unroll
      for (int r = 0; r < 8; ++r) {
        const float* brow = bt + (r + hlf * 8) * BPITCH + h;
        s0[r] += brow[col * 8];
        s1[r] += brow[(col + 16) * 8];
      }
    }

    // ---- online softmax update (row reductions across 16-lane half-groups)
#pragma unroll
    for (int r = 0; r < 8; ++r) {
      float mx = fmaxf(s0[r], s1[r]);
      mx = fmaxf(mx, __shfl_xor(mx, 1, 32));
      mx = fmaxf(mx, __shfl_xor(mx, 2, 32));
      mx = fmaxf(mx, __shfl_xor(mx, 4, 32));
      mx = fmaxf(mx, __shfl_xor(mx, 8, 32));
      const float mnew = fmaxf(mrun[r], mx);
      const float p0 = __expf(s0[r] - mnew);
      const float p1 = __expf(s1[r] - mnew);
      pshare[h][(r + hlf * 8) * PPITCH + col]      = f2bf(p0);
      pshare[h][(r + hlf * 8) * PPITCH + 16 + col] = f2bf(p1);
      float rs = p0 + p1;
      rs += __shfl_xor(rs, 1, 32);
      rs += __shfl_xor(rs, 2, 32);
      rs += __shfl_xor(rs, 4, 32);
      rs += __shfl_xor(rs, 8, 32);
      const float cor = __expf(mrun[r] - mnew);
      lrun[r] = lrun[r] * cor + rs;
      mrun[r] = mnew;
#pragma unroll
      for (int t = 0; t < 4; ++t) oacc[t][r] *= cor;
    }

    // ---- read P back in A-matrix layout (same-wave LDS: DS ops are in-order)
    Frag ap;
    const unsigned short* prow = &pshare[h][0] + (size_t)col * PPITCH + hlf * 8;
    ap.q[0] = *(const uint4*)(prow);
    ap.q[1] = *(const uint4*)(prow + 16);

    // ---- O += P @ V over 4 dh column tiles
#pragma unroll
    for (int t = 0; t < 4; ++t) {
      Frag bv;
      const unsigned short* vrow = vh + (size_t)(t * 16 + col) * NN + j0 + hlf * 8;
      bv.q[0] = *(const uint4*)(vrow);
      bv.q[1] = *(const uint4*)(vrow + 16);
      oacc[t] = __builtin_amdgcn_wmma_f32_16x16x32_bf16(
          false, ap.v, false, bv.v, (short)0, oacc[t], false, false);
    }

    stage_wait();  // next bias buffer ready; current one free for restage
    cur ^= 1;
  }

  // ---- finalize: O / l, store bf16 [B,N,D] for the output projection
#pragma unroll
  for (int r = 0; r < 8; ++r) {
    const float inv = 1.0f / lrun[r];
    const int i = i0 + r + hlf * 8;
#pragma unroll
    for (int t = 0; t < 4; ++t) {
      aob[((size_t)(b * NN + i)) * DDIM + h * DH + t * 16 + col] =
          f2bf(oacc[t][r] * inv);
    }
  }
}

// ---------------------------------------------------------------- out proj
__global__ __launch_bounds__(32) void proj_gemm_kernel(
    const unsigned short* __restrict__ aob,
    const unsigned short* __restrict__ wmb,
    const float* __restrict__ bm,
    float* __restrict__ out) {
  const int m0 = blockIdx.x * 16;
  const int n0 = blockIdx.y * 64;
  const int lane = threadIdx.x;
  const int col  = lane & 15;
  const int hlf  = lane >> 4;

  const unsigned short* arow  = aob + (size_t)(m0 + col) * DDIM + hlf * 8;
  const unsigned short* wrow0 = wmb + (size_t)(n0 + col) * DDIM + hlf * 8;

  v8f acc[4];
#pragma unroll
  for (int t = 0; t < 4; ++t)
#pragma unroll
    for (int r = 0; r < 8; ++r) acc[t][r] = 0.0f;

  Frag a[2], b[2][4];
  auto loadA = [&](Frag& f, int k0) {
    f.q[0] = *(const uint4*)(arow + k0);
    f.q[1] = *(const uint4*)(arow + k0 + 16);
  };
  auto loadB = [&](Frag* f, int k0) {
#pragma unroll
    for (int t = 0; t < 4; ++t) {
      const unsigned short* wr = wrow0 + (size_t)t * 16 * DDIM + k0;
      f[t].q[0] = *(const uint4*)(wr);
      f[t].q[1] = *(const uint4*)(wr + 16);
    }
  };

  loadA(a[0], 0);
  loadB(b[0], 0);
  for (int k0 = 0; k0 < DDIM; k0 += 64) {
    loadA(a[1], k0 + 32);
    loadB(b[1], k0 + 32);
#pragma unroll
    for (int t = 0; t < 4; ++t)
      acc[t] = __builtin_amdgcn_wmma_f32_16x16x32_bf16(
          false, a[0].v, false, b[0][t].v, (short)0, acc[t], false, false);
    if (k0 + 64 < DDIM) {
      loadA(a[0], k0 + 64);
      loadB(b[0], k0 + 64);
    }
#pragma unroll
    for (int t = 0; t < 4; ++t)
      acc[t] = __builtin_amdgcn_wmma_f32_16x16x32_bf16(
          false, a[1].v, false, b[1][t].v, (short)0, acc[t], false, false);
  }

#pragma unroll
  for (int t = 0; t < 4; ++t) {
#pragma unroll
    for (int r = 0; r < 8; ++r) {
      const int i = m0 + r + hlf * 8;
      const int j = n0 + t * 16 + col;
      out[(size_t)i * DDIM + j] = acc[t][r] + bm[j];
    }
  }
}

// ---------------------------------------------------------------- launch
extern "C" void kernel_launch(void* const* d_in, const int* in_sizes, int n_in,
                              void* d_out, int out_size, void* d_ws, size_t ws_size,
                              hipStream_t stream) {
  const float* x    = (const float*)d_in[0];
  const float* bias = (const float*)d_in[1];
  const float* Wq   = (const float*)d_in[2];
  const float* Wk   = (const float*)d_in[3];
  const float* Wv   = (const float*)d_in[4];
  const float* Wm   = (const float*)d_in[5];
  const float* bm   = (const float*)d_in[6];
  float* out = (float*)d_out;

  const size_t X   = (size_t)BB * NN * DDIM;  // 4,194,304 elements
  const size_t WSZ = (size_t)DDIM * DDIM;     //   262,144 elements

  // Workspace carve-up (bf16 buffers), ~42 MiB total.
  char* p = (char*)d_ws;
  unsigned short* xb  = (unsigned short*)p; p += X * 2;
  unsigned short* wqb = (unsigned short*)p; p += WSZ * 2;
  unsigned short* wkb = (unsigned short*)p; p += WSZ * 2;
  unsigned short* wvb = (unsigned short*)p; p += WSZ * 2;
  unsigned short* wmb = (unsigned short*)p; p += WSZ * 2;
  unsigned short* qb  = (unsigned short*)p; p += X * 2;
  unsigned short* kb  = (unsigned short*)p; p += X * 2;
  unsigned short* vb  = (unsigned short*)p; p += X * 2;
  unsigned short* aob = (unsigned short*)p; p += X * 2;

  f32_to_bf16_kernel<<<2048, 256, 0, stream>>>(x,  xb,  (int)X);
  f32_to_bf16_kernel<<<256,  256, 0, stream>>>(Wq, wqb, (int)WSZ);
  f32_to_bf16_kernel<<<256,  256, 0, stream>>>(Wk, wkb, (int)WSZ);
  f32_to_bf16_kernel<<<256,  256, 0, stream>>>(Wv, wvb, (int)WSZ);
  f32_to_bf16_kernel<<<256,  256, 0, stream>>>(Wm, wmb, (int)WSZ);

  qkv_gemm_kernel<<<dim3((BB * NN) / 16, DDIM / 64, 3), 32, 0, stream>>>(
      xb, wqb, wkb, wvb, qb, kb, vb);

  attn_kernel<<<dim3(NN / 16, BB), dim3(32, HH), 0, stream>>>(
      qb, kb, vb, bias, aob);

  proj_gemm_kernel<<<dim3((BB * NN) / 16, DDIM / 64), 32, 0, stream>>>(
      aob, wmb, bm, out);
}
